// ThirdOrderAttention_71116068487403
// MI455X (gfx1250) — matrix operand order repaired and checked by reading, not compiled, not run on hardware
//
#include <hip/hip_runtime.h>

// ---------------------------------------------------------------------------
// CDNA5 (gfx1250, wave32) fused third-order attention.
//   B=1, N=256, C=512, H=8, D=64.
// Per (head h, query i) workgroup:
//   S_i = ((kj .* q_i) @ kk^T) * scale      -> 256x256, kept in LDS (bf16)
//   P_i = softmax(S_i flat)
//   y_i = sum_j vj[j,:] .* (P_i @ vk)[j,:]
// Matmuls use v_wmma_f32_16x16x32_bf16; the per-head 256x256 bf16 K/V tile is
// staged into LDS with one Tensor Data Mover descriptor (tensor_load_to_lds),
// synchronized with s_wait_tensorcnt.
// ---------------------------------------------------------------------------

typedef __bf16 bf16_t;
typedef __attribute__((ext_vector_type(16))) __bf16 v16bf;
typedef __attribute__((ext_vector_type(8)))  __bf16 v8bf;
typedef __attribute__((ext_vector_type(8)))  float  v8f;
typedef __attribute__((ext_vector_type(4)))  unsigned int u32x4;
typedef __attribute__((ext_vector_type(8)))  int i32x8;
typedef __attribute__((ext_vector_type(4)))  int i32x4;

#define NTOK 256
#define DHEAD 64
#define CDIM 512
#define NHEAD 8
#define PROJW 2560   // 5*C

__device__ inline v8f wmma_bf16(v16bf a, v16bf b, v8f c) {
  // (neg_a, A, neg_b, B, c_mod, C, reuse_a, reuse_b)
  return __builtin_amdgcn_wmma_f32_16x16x32_bf16(false, a, false, b, (short)0, c,
                                                 false, false);
}

// A fragment: 16x32 (MxK) bf16, row-major source, row stride ld (elements).
// Lane L<16 holds row M=L; VGPR0..3 -> K=0..7, VGPR4..7 -> K=16..23.
// Lane L>=16 holds row M=L-16; K=8..15 and K=24..31.
__device__ inline v16bf load_a_frag(const bf16_t* src, int ld, int lane) {
  int row = lane & 15;
  int kh  = (lane >> 4) & 1;
  const bf16_t* p = src + row * ld + kh * 8;
  v16bf a;
#pragma unroll
  for (int e = 0; e < 8; ++e) a[e] = p[e];
#pragma unroll
  for (int e = 0; e < 8; ++e) a[8 + e] = p[16 + e];
  return a;
}

// B fragment: 32x16 (KxN) bf16 from an N-major source: src[n*ld + k].
// Lane L: column N = L&15; elements e<8 -> K = (L>>4)*8 + e, e>=8 -> K = 16 + (L>>4)*8 + (e-8).
__device__ inline v16bf load_b_frag_nk(const bf16_t* src, int ld, int lane) {
  int col = lane & 15;
  int kh  = (lane >> 4) & 1;
  const bf16_t* p = src + col * ld + kh * 8;
  v16bf b;
#pragma unroll
  for (int e = 0; e < 8; ++e) b[e] = p[e];
#pragma unroll
  for (int e = 0; e < 8; ++e) b[8 + e] = p[16 + e];
  return b;
}

// ---------------------------------------------------------------------------
// f32 -> bf16 conversion (flat)
// ---------------------------------------------------------------------------
__global__ void cvt_bf16(const float* __restrict__ in, bf16_t* __restrict__ out, int n) {
  int i = blockIdx.x * blockDim.x + threadIdx.x;
  if (i < n) out[i] = (bf16_t)in[i];
}

// f32 [rows][cols] -> bf16 transposed [cols][rows]
__global__ void cvt_bf16_t(const float* __restrict__ in, bf16_t* __restrict__ out,
                           int rows, int cols) {
  int i = blockIdx.x * blockDim.x + threadIdx.x;
  if (i < rows * cols) {
    int r = i / cols, c = i - r * cols;
    out[c * rows + r] = (bf16_t)in[i];
  }
}

// ---------------------------------------------------------------------------
// Generic WMMA GEMM: C[M,N] = A[M,K] @ B[K,N] + bias[N]
// A row-major bf16, BT is B stored N-major ([n][k]) bf16.
// One wave per 16x16 tile, 4 waves per block.
// ---------------------------------------------------------------------------
__global__ __launch_bounds__(128) void gemm_bf16(
    const bf16_t* __restrict__ A, const bf16_t* __restrict__ BT,
    const float* __restrict__ bias, bf16_t* __restrict__ Cb,
    float* __restrict__ Cf, int M, int Nn, int K) {
  int tile = blockIdx.x * 4 + (threadIdx.x >> 5);
  int lane = threadIdx.x & 31;
  int tilesN = Nn >> 4;
  int mt = tile / tilesN;
  int nt = tile - mt * tilesN;
  if (mt * 16 >= M) return;

  v8f c = {0.f, 0.f, 0.f, 0.f, 0.f, 0.f, 0.f, 0.f};
  const bf16_t* arow = A + mt * 16 * K;
  const bf16_t* brow = BT + nt * 16 * K;
  for (int ks = 0; ks < K; ks += 32) {
    __builtin_prefetch(arow + ks + 64, 0, 0);
    v16bf a = load_a_frag(arow + ks, K, lane);
    v16bf b = load_b_frag_nk(brow + ks, K, lane);
    c = wmma_bf16(a, b, c);
  }

  int col = lane & 15;
  int mh  = ((lane >> 4) & 1) * 8;
#pragma unroll
  for (int r = 0; r < 8; ++r) {
    int m = mt * 16 + mh + r;
    int n = nt * 16 + col;
    float v = c[r] + bias[n];
    if (Cb) Cb[m * Nn + n] = (bf16_t)v;
    if (Cf) Cf[m * Nn + n] = v;
  }
}

// ---------------------------------------------------------------------------
// Fused third-order attention. grid = (N, H), block = 256 (8 waves).
// P5: [N][2560] bf16 projection (q|kj|kk|vj|vk per head, 64 each).
// Yb: [N][C] bf16 attention output (input to the final projection GEMM).
//
// LDS layout:
//   sKV [256][256] bf16 : head-h columns 64..319 of P5 (kj|kk|vj|vk) -- one
//                         2D TDM tile (row stride 2560 elements in memory)
//   sVkT [64][256] bf16 : vk transposed (for contiguous B-fragment loads)
//   sS  [256][256] bf16 : scores -> probabilities
// ---------------------------------------------------------------------------
__global__ __launch_bounds__(256) void attn_kernel(
    const bf16_t* __restrict__ P5, bf16_t* __restrict__ Yb) {
  extern __shared__ char smem_raw[];
  bf16_t* sKV  = (bf16_t*)smem_raw;           // [256][256]
  bf16_t* sVkT = sKV  + NTOK * NTOK;          // [64][256]
  bf16_t* sS   = sVkT + DHEAD * NTOK;         // [256][256]
  float*  sQ   = (float*)(sS + NTOK * NTOK);  // [64]
  float*  yacc = sQ + DHEAD;                  // [64]
  float*  red  = yacc + DHEAD;                // [8]
  float*  sval = red + 8;                     // [0]=max [1]=sum

  const int tid  = threadIdx.x;
  const int w    = tid >> 5;
  const int lane = tid & 31;
  const int row  = lane & 15;
  const int col  = lane & 15;
  const int kh   = (lane >> 4) & 1;
  const int mh   = kh * 8;
  const int h    = blockIdx.y;
  const int qi   = blockIdx.x;
  const float scale = 0.125f;  // D^-0.5 = 64^-0.5

  const bf16_t* base = P5 + h * (5 * DHEAD);

  // ---- stage the 256x256 bf16 K/V tile for this head into LDS ---------
#if __has_builtin(__builtin_amdgcn_tensor_load_to_lds)
  if (tid < 32) {
    if (lane == 0) {
      // Tensor DMA descriptor (D#), ISA 8.3/8.4. 2D tile, data_size=2B,
      // tile 256x256, tensor row stride = 2560 elements.
      unsigned long long ga = (unsigned long long)(base + DHEAD);  // tile start
      unsigned int lds_off  = (unsigned int)(unsigned long long)sKV;
      u32x4 g0;
      g0[0] = 1u;                                   // count=1, user mode
      g0[1] = lds_off;                              // lds_addr (bytes)
      g0[2] = (unsigned int)(ga & 0xffffffffu);     // global_addr[31:0]
      g0[3] = (unsigned int)((ga >> 32) & 0x1ffffffu) | (2u << 30);  // type=2
      i32x8 g1;
      g1[0] = 1 << 16;            // workgroup_mask=0, data_size=1 (2 bytes)
      g1[1] = (int)(256u << 16);  // tensor_dim0[15:0] = 256 (bits 48..63)
      g1[2] = (int)(256u << 16);  // tensor_dim0 hi=0, tensor_dim1 lo = 256
      g1[3] = (int)(256u << 16);  // tensor_dim1 hi=0, tile_dim0 = 256
      g1[4] = 256;                // tile_dim1 = 256, tile_dim2 = 0
      g1[5] = 2560;               // tensor_dim0_stride[31:0] = 2560
      g1[6] = 0;                  // stride hi / dim1_stride lo
      g1[7] = 0;
      i32x4 gz4 = {0, 0, 0, 0};             // groups 2/3 unused (2D tile)
      i32x8 gz8 = {0, 0, 0, 0, 0, 0, 0, 0}; // extra group (6-arg form)
      __builtin_amdgcn_tensor_load_to_lds(g0, g1, gz4, gz4, gz8, 0);
    }
#if __has_builtin(__builtin_amdgcn_s_wait_tensorcnt)
    __builtin_amdgcn_s_wait_tensorcnt(0);
#endif
  }
#else
  // Fallback: vectorized 16-byte copies (256 rows x 128 bf16).
  for (int c = tid; c < NTOK * 16; c += 256) {
    int n = c >> 4, part = c & 15;
    *(v8bf*)(sKV + n * NTOK + part * 8) =
        *(const v8bf*)(base + n * PROJW + DHEAD + part * 8);
  }
#endif
  if (tid < DHEAD) {
    sQ[tid]   = (float)base[qi * PROJW + tid] * scale;
    yacc[tid] = 0.f;
  }
  __syncthreads();

  // ---- transpose vk (sKV cols 192..255) into sVkT[64][256] ------------
  for (int t = tid; t < DHEAD * NTOK; t += 256) {
    int d = t >> 8, n = t & 255;
    sVkT[d * NTOK + n] = sKV[n * NTOK + 192 + d];
  }

  // per-lane q values laid out to match the A-fragment element mapping
  float q0[16], q1[16];
#pragma unroll
  for (int e = 0; e < 16; ++e) {
    int dk = (e < 8) ? (kh * 8 + e) : (16 + kh * 8 + (e - 8));
    q0[e] = sQ[dk];
    q1[e] = sQ[32 + dk];
  }

  // ---- GEMM1: S = (kj .* q_i) @ kk^T ; track max ----------------------
  float lmax = -3.0e38f;
  for (int jt = w; jt < 16; jt += 8) {
    const bf16_t* kr = sKV + (jt * 16 + row) * NTOK;  // kj = cols 0..63
    v16bf a0, a1;
#pragma unroll
    for (int e = 0; e < 16; ++e) {
      int dk = (e < 8) ? (kh * 8 + e) : (16 + kh * 8 + (e - 8));
      a0[e] = (bf16_t)((float)kr[dk] * q0[e]);
      a1[e] = (bf16_t)((float)kr[32 + dk] * q1[e]);
    }
    for (int kt = 0; kt < 16; ++kt) {
      // kk = cols 64..127 of sKV
      v16bf b0 = load_b_frag_nk(sKV + kt * 16 * NTOK + 64, NTOK, lane);  // d 0..31
      v16bf b1 = load_b_frag_nk(sKV + kt * 16 * NTOK + 96, NTOK, lane);  // d 32..63
      v8f c = {0.f, 0.f, 0.f, 0.f, 0.f, 0.f, 0.f, 0.f};
      c = wmma_bf16(a0, b0, c);
      c = wmma_bf16(a1, b1, c);
#pragma unroll
      for (int r = 0; r < 8; ++r) {
        float v = c[r];
        lmax = fmaxf(lmax, v);
        sS[(jt * 16 + mh + r) * NTOK + kt * 16 + col] = (bf16_t)v;
      }
    }
  }
#pragma unroll
  for (int o = 16; o > 0; o >>= 1) lmax = fmaxf(lmax, __shfl_xor(lmax, o));
  if (lane == 0) red[w] = lmax;
  __syncthreads();
  if (tid == 0) {
    float m = red[0];
    for (int p = 1; p < 8; ++p) m = fmaxf(m, red[p]);
    sval[0] = m;
  }
  __syncthreads();
  const float gmax = sval[0];

  // ---- softmax over the flat 256*256 plane ----------------------------
  float lsum = 0.f;
  for (int idx = tid; idx < NTOK * NTOK; idx += 256) {
    float s = (float)sS[idx];
    float p = __expf(s - gmax);
    lsum += p;
    sS[idx] = (bf16_t)p;
  }
#pragma unroll
  for (int o = 16; o > 0; o >>= 1) lsum += __shfl_xor(lsum, o);
  if (lane == 0) red[w] = lsum;
  __syncthreads();
  if (tid == 0) {
    float s = 0.f;
    for (int p = 0; p < 8; ++p) s += red[p];
    sval[1] = s;
  }
  __syncthreads();
  const float inv = 1.f / sval[1];

  // ---- GEMM2: U = P @ vk ; y += colsum(U .* vj) -----------------------
  for (int t = w; t < 64; t += 8) {
    int jt = t >> 2;
    int dt = t & 3;
    v8f c = {0.f, 0.f, 0.f, 0.f, 0.f, 0.f, 0.f, 0.f};
    for (int ks = 0; ks < 8; ++ks) {
      v16bf a = load_a_frag(sS + jt * 16 * NTOK + ks * 32, NTOK, lane);
      v16bf b = load_b_frag_nk(sVkT + dt * 16 * NTOK + ks * 32, NTOK, lane);
      c = wmma_bf16(a, b, c);
    }
    int n = dt * 16 + col;
    float partial = 0.f;
#pragma unroll
    for (int r = 0; r < 8; ++r)
      partial += c[r] * (float)sKV[(jt * 16 + mh + r) * NTOK + 128 + n];  // vj
    atomicAdd(&yacc[n], partial);  // ds_add_f32
  }
  __syncthreads();

  if (tid < DHEAD)
    Yb[qi * CDIM + h * DHEAD + tid] = (bf16_t)(yacc[tid] * inv);
}

// ---------------------------------------------------------------------------
// Host side
// ---------------------------------------------------------------------------
extern "C" void kernel_launch(void* const* d_in, const int* in_sizes, int n_in,
                              void* d_out, int out_size, void* d_ws, size_t ws_size,
                              hipStream_t stream) {
  const float* x     = (const float*)d_in[0];  // [256][512]
  const float* W_att = (const float*)d_in[1];  // [512][2560]
  const float* b_att = (const float*)d_in[2];  // [2560]
  const float* W_out = (const float*)d_in[3];  // [512][512]
  const float* b_out = (const float*)d_in[4];  // [512]
  float* out = (float*)d_out;                  // [256][512]

  // workspace carve-out (all offsets 256B aligned)
  char* ws = (char*)d_ws;
  size_t off = 0;
  auto carve = [&](size_t bytes) {
    void* p = ws + off;
    off += (bytes + 255) & ~size_t(255);
    return p;
  };
  bf16_t* xb  = (bf16_t*)carve((size_t)NTOK * CDIM * 2);       // x bf16
  bf16_t* WaT = (bf16_t*)carve((size_t)PROJW * CDIM * 2);      // W_att^T bf16 [2560][512]
  bf16_t* P5  = (bf16_t*)carve((size_t)NTOK * PROJW * 2);      // projection bf16
  bf16_t* Yb  = (bf16_t*)carve((size_t)NTOK * CDIM * 2);       // attention out bf16
  bf16_t* WoT = (bf16_t*)carve((size_t)CDIM * CDIM * 2);       // W_out^T bf16 [512][512]

  // 1) conversions (transpose the B operands to N-major for WMMA B-frags)
  {
    int n = NTOK * CDIM;
    cvt_bf16<<<(n + 255) / 256, 256, 0, stream>>>(x, xb, n);
  }
  {
    int n = CDIM * PROJW;
    cvt_bf16_t<<<(n + 255) / 256, 256, 0, stream>>>(W_att, WaT, CDIM, PROJW);
  }
  {
    int n = CDIM * CDIM;
    cvt_bf16_t<<<(n + 255) / 256, 256, 0, stream>>>(W_out, WoT, CDIM, CDIM);
  }

  // 2) projection: P5[256][2560] = x @ W_att + b_att   (bf16 out)
  {
    int tiles = (NTOK / 16) * (PROJW / 16);  // 16*160 = 2560
    gemm_bf16<<<tiles / 4, 128, 0, stream>>>(xb, WaT, b_att, P5, nullptr,
                                             NTOK, PROJW, CDIM);
  }

  // 3) fused third-order attention: one workgroup per (query i, head h)
  {
    size_t smem = (size_t)(2 * NTOK * NTOK + DHEAD * NTOK) * sizeof(bf16_t) +
                  (size_t)(2 * DHEAD + 8 + 2) * sizeof(float);  // ~289 KB < 320 KB
    attn_kernel<<<dim3(NTOK, NHEAD), 256, smem, stream>>>(P5, Yb);
  }

  // 4) output projection: out[256][512] = Y @ W_out + b_out   (f32 out)
  {
    int tiles = (NTOK / 16) * (CDIM / 16);  // 16*32 = 512
    gemm_bf16<<<tiles / 4, 128, 0, stream>>>(Yb, WoT, b_out, nullptr, out,
                                             NTOK, CDIM, CDIM);
  }
}
